// TannerGNN_22677427323113
// MI455X (gfx1250) — matrix operand order repaired and checked by reading, not compiled, not run on hardware
//
#include <hip/hip_runtime.h>

#define H 64
#define NLAYERS 3
#define ETYPES 2

typedef __attribute__((ext_vector_type(16))) _Float16 v16h;
typedef __attribute__((ext_vector_type(8)))  float    v8f;

// ---------------------------------------------------------------------------
// WMMA fragment helpers (CDNA5 wave32, V_WMMA_F32_16X16X32_F16)
//
// A-matrix 16x32 f16 layout (ISA 7.12.2): lane = g*16 + m, row M=m.
//   element e of v16h: v=e>>1, K = (v<4?0:16) + g*8 + 2*(v&3) + (e&1)
// B-matrix 32x16 f16 layout: lane = g*16 + n, col N=n, element e -> K = (e>>3)*16 + g*8 + (e&7).
//   (weights are pre-packed so each lane's 16 halves are contiguous)
// C/D 16x16 f32 layout: lane l, vgpr r -> (M = (l>>4)*8 + r, N = l&15)
// ---------------------------------------------------------------------------

__device__ __forceinline__ int a_k0(int v, int g) {
  return ((v < 4) ? 0 : 16) + (g << 3) + ((v & 3) << 1);
}

__device__ __forceinline__ v16h load_a_frag_h(const _Float16* base, int g) {
  v16h a;
#pragma unroll
  for (int v = 0; v < 8; ++v) {
    int k0 = a_k0(v, g);
    union { unsigned int u; _Float16 h[2]; } cv;
    cv.u = *reinterpret_cast<const unsigned int*>(base + k0);
    a[2 * v]     = cv.h[0];
    a[2 * v + 1] = cv.h[1];
  }
  return a;
}

__device__ __forceinline__ v16h load_a_frag_f(const float* base, int g) {
  v16h a;
#pragma unroll
  for (int v = 0; v < 8; ++v) {
    int k0 = a_k0(v, g);
    a[2 * v]     = (_Float16)base[k0];
    a[2 * v + 1] = (_Float16)base[k0 + 1];
  }
  return a;
}

__device__ __forceinline__ v8f wmma16(v16h a, v16h b, v8f c) {
  return __builtin_amdgcn_wmma_f32_16x16x32_f16(false, a, false, b, (short)0, c,
                                                false, false);
}

__device__ __forceinline__ v8f vzero8() {
  v8f z = {0.f, 0.f, 0.f, 0.f, 0.f, 0.f, 0.f, 0.f};
  return z;
}

// ---------------------------------------------------------------------------
// Weight repack: row-major f32 (nmat, K, N) -> f16 B-fragments
// dst layout: [mat][kc = K/32][nt = N/16][lane 32][elem 16]
// ---------------------------------------------------------------------------
__global__ void pack_b_kernel(const float* __restrict__ src,
                              _Float16* __restrict__ dst,
                              int K, int N, int total) {
  int t = blockIdx.x * blockDim.x + threadIdx.x;
  if (t >= total) return;
  int e    = t & 15;
  int lane = (t >> 4) & 31;
  int rest = t >> 9;
  int ntc  = N >> 4;
  int kcnt = K >> 5;
  int nt   = rest % ntc; rest /= ntc;
  int kc   = rest % kcnt;
  int mat  = rest / kcnt;
  int k = (kc << 5) + ((e >> 3) << 4) + ((lane >> 4) << 3) + (e & 7);
  int n = (nt << 4) + (lane & 15);
  dst[t] = (_Float16)src[((size_t)mat * K + k) * N + n];
}

// ---------------------------------------------------------------------------
// Input projection: h = relu(LayerNorm(x @ in_w + in_b)); one wave per node.
// Also materializes the f16 copy of h for WMMA gathers.
// ---------------------------------------------------------------------------
__global__ void input_proj_kernel(const float* __restrict__ x,
                                  const float* __restrict__ in_w,
                                  const float* __restrict__ in_b,
                                  const float* __restrict__ ln_g,
                                  const float* __restrict__ ln_b,
                                  float* __restrict__ h,
                                  _Float16* __restrict__ h_half, int n) {
  int gid  = blockIdx.x * blockDim.x + threadIdx.x;
  int node = gid >> 5, lane = gid & 31;
  if (node >= n) return;
  float x0 = x[node * 4 + 0], x1 = x[node * 4 + 1];
  float x2 = x[node * 4 + 2], x3 = x[node * 4 + 3];
  int c0 = lane * 2, c1 = c0 + 1;
  float y0 = in_b[c0] + x0 * in_w[c0] + x1 * in_w[64 + c0] +
             x2 * in_w[128 + c0] + x3 * in_w[192 + c0];
  float y1 = in_b[c1] + x0 * in_w[c1] + x1 * in_w[64 + c1] +
             x2 * in_w[128 + c1] + x3 * in_w[192 + c1];
  float s = y0 + y1;
#pragma unroll
  for (int off = 16; off > 0; off >>= 1) s += __shfl_xor(s, off, 32);
  float mean = s * (1.0f / 64.0f);
  float d0 = y0 - mean, d1 = y1 - mean;
  float q = d0 * d0 + d1 * d1;
#pragma unroll
  for (int off = 16; off > 0; off >>= 1) q += __shfl_xor(q, off, 32);
  float rs = rsqrtf(q * (1.0f / 64.0f) + 1e-5f);
  float z0 = fmaxf(d0 * rs * ln_g[c0] + ln_b[c0], 0.0f);
  float z1 = fmaxf(d1 * rs * ln_g[c1] + ln_b[c1], 0.0f);
  size_t base = (size_t)node * H;
  h[base + c0] = z0;
  h[base + c1] = z1;
  h_half[base + c0] = (_Float16)z0;
  h_half[base + c1] = (_Float16)z1;
}

// ---------------------------------------------------------------------------
// Fused edge-message kernel: one wave per 16-edge tile.
//   mi = [h[src], h[dst]]  (16 x 128, gathered as f16 A-fragments)
//   for e in {0,1}: out = relu(mi @ W1[l,e] + b1) @ W2[l,e] + b2
//   atomicAdd into agg[dst] for rows whose edge_type == e
// ---------------------------------------------------------------------------
__global__ void __launch_bounds__(256) msg_kernel(
    const _Float16* __restrict__ h_half,
    const int* __restrict__ esrc, const int* __restrict__ edst,
    const int* __restrict__ etype,
    const _Float16* __restrict__ w1p, const _Float16* __restrict__ w2p,
    const float* __restrict__ b1, const float* __restrict__ b2,
    float* __restrict__ agg, int layer, int ntiles) {
  __shared__ _Float16 lds_hid[8][16][H];  // per-wave 16x64 hidden tile
  int lane = threadIdx.x & 31;
  int w    = threadIdx.x >> 5;
  int tile = blockIdx.x * 8 + w;
  if (tile >= ntiles) return;
  int g = lane >> 4, m = lane & 15;
  int ebase = tile * 16;

  int sm = esrc[ebase + m];
  int dm = edst[ebase + m];
  const _Float16* hs = h_half + (size_t)sm * H;
  const _Float16* hd = h_half + (size_t)dm * H;
  v16h a0 = load_a_frag_h(hs, g);        // K   0..31  (src)
  v16h a1 = load_a_frag_h(hs + 32, g);   // K  32..63  (src)
  v16h a2 = load_a_frag_h(hd, g);        // K  64..95  (dst)
  v16h a3 = load_a_frag_h(hd + 32, g);   // K  96..127 (dst)

  int etM[8], dnM[8];                    // per C-layout row M = g*8 + r
#pragma unroll
  for (int r = 0; r < 8; ++r) {
    etM[r] = etype[ebase + g * 8 + r];
    dnM[r] = edst[ebase + g * 8 + r];
  }

  for (int e = 0; e < ETYPES; ++e) {
    // ---- MLP layer 1: (16x128) @ (128x64) ----
    const _Float16* w1base = w1p + (size_t)(layer * 2 + e) * 16 * 512;
    v8f c1[4];
    c1[0] = c1[1] = c1[2] = c1[3] = vzero8();
#pragma unroll
    for (int kc = 0; kc < 4; ++kc) {
      v16h a = (kc == 0) ? a0 : (kc == 1) ? a1 : (kc == 2) ? a2 : a3;
#pragma unroll
      for (int nt = 0; nt < 4; ++nt) {
        v16h b = *(const v16h*)(w1base + (kc * 4 + nt) * 512 + lane * 16);
        c1[nt] = wmma16(a, b, c1[nt]);
      }
    }
    // bias + ReLU, re-layout C-frag -> A-frag through LDS (wave-local)
    const float* b1v = b1 + (layer * 2 + e) * H;
#pragma unroll
    for (int nt = 0; nt < 4; ++nt) {
      float bb = b1v[nt * 16 + m];
#pragma unroll
      for (int r = 0; r < 8; ++r) {
        float vv = c1[nt][r] + bb;
        lds_hid[w][g * 8 + r][nt * 16 + m] = (_Float16)fmaxf(vv, 0.0f);
      }
    }
    asm volatile("s_wait_dscnt 0" ::: "memory");  // drain wave's DS stores

    // ---- MLP layer 2: (16x64) @ (64x64) ----
    const _Float16* w2base = w2p + (size_t)(layer * 2 + e) * 8 * 512;
    v8f c2[4];
    c2[0] = c2[1] = c2[2] = c2[3] = vzero8();
#pragma unroll
    for (int kc = 0; kc < 2; ++kc) {
      v16h a = load_a_frag_h(&lds_hid[w][m][kc * 32], g);
#pragma unroll
      for (int nt = 0; nt < 4; ++nt) {
        v16h b = *(const v16h*)(w2base + (kc * 4 + nt) * 512 + lane * 16);
        c2[nt] = wmma16(a, b, c2[nt]);
      }
    }
    // ---- scatter-add (segment sum) where edge_type matches ----
    const float* b2v = b2 + (layer * 2 + e) * H;
#pragma unroll
    for (int nt = 0; nt < 4; ++nt) {
      float bb = b2v[nt * 16 + m];
#pragma unroll
      for (int r = 0; r < 8; ++r) {
        if (etM[r] == e) {
          atomicAdd(&agg[(size_t)dnM[r] * H + nt * 16 + m], c2[nt][r] + bb);
        }
      }
    }
  }
}

// ---------------------------------------------------------------------------
// GRU cell: one wave per 16-node tile.
//   gi = agg @ wi + bi ; gh = h @ wh + bh ; gates [r, z, n] (torch order)
// ---------------------------------------------------------------------------
__global__ void __launch_bounds__(256) gru_kernel(
    float* __restrict__ h, _Float16* __restrict__ h_half,
    const float* __restrict__ agg,
    const _Float16* __restrict__ wip, const _Float16* __restrict__ whp,
    const float* __restrict__ gbi, const float* __restrict__ gbh,
    int layer, int ntiles) {
  int gid  = blockIdx.x * blockDim.x + threadIdx.x;
  int tile = gid >> 5, lane = gid & 31;
  if (tile >= ntiles) return;
  int g = lane >> 4, m = lane & 15;
  int nbase = tile * 16;

  const float* ar = agg + (size_t)(nbase + m) * H;
  v16h Aa0 = load_a_frag_f(ar, g);
  v16h Aa1 = load_a_frag_f(ar + 32, g);
  const _Float16* hr = h_half + (size_t)(nbase + m) * H;
  v16h Ah0 = load_a_frag_h(hr, g);
  v16h Ah1 = load_a_frag_h(hr + 32, g);

  const _Float16* wib = wip + (size_t)layer * 24 * 512;  // [kc 2][nt 12][512]
  const _Float16* whb = whp + (size_t)layer * 24 * 512;
  const float* bi = gbi + layer * 192;
  const float* bh = gbh + layer * 192;

  auto gate_mm = [&](const _Float16* wb, v16h A0, v16h A1, int ntoff,
                     v8f out[4]) {
#pragma unroll
    for (int nt = 0; nt < 4; ++nt) {
      v16h b0 = *(const v16h*)(wb + (size_t)(0 * 12 + ntoff + nt) * 512 + lane * 16);
      out[nt] = wmma16(A0, b0, out[nt]);
      v16h b1 = *(const v16h*)(wb + (size_t)(1 * 12 + ntoff + nt) * 512 + lane * 16);
      out[nt] = wmma16(A1, b1, out[nt]);
    }
  };

  v8f R[4], Z[4], In[4], Hn[4];
  R[0] = R[1] = R[2] = R[3] = vzero8();
  Z[0] = Z[1] = Z[2] = Z[3] = vzero8();
  In[0] = In[1] = In[2] = In[3] = vzero8();
  Hn[0] = Hn[1] = Hn[2] = Hn[3] = vzero8();
  gate_mm(wib, Aa0, Aa1, 0, R);   // i_r
  gate_mm(whb, Ah0, Ah1, 0, R);   // + h_r
  gate_mm(wib, Aa0, Aa1, 4, Z);   // i_z
  gate_mm(whb, Ah0, Ah1, 4, Z);   // + h_z
  gate_mm(wib, Aa0, Aa1, 8, In);  // i_n
  gate_mm(whb, Ah0, Ah1, 8, Hn);  // h_n

#pragma unroll
  for (int nt = 0; nt < 4; ++nt) {
    int col = nt * 16 + m;
    float bir = bi[col],       bhr = bh[col];
    float biz = bi[64 + col],  bhz = bh[64 + col];
    float bin = bi[128 + col], bhn = bh[128 + col];
#pragma unroll
    for (int r = 0; r < 8; ++r) {
      int M = g * 8 + r;
      size_t idx = (size_t)(nbase + M) * H + col;
      float rr = 1.0f / (1.0f + __expf(-(R[nt][r] + bir + bhr)));
      float zz = 1.0f / (1.0f + __expf(-(Z[nt][r] + biz + bhz)));
      float nn = tanhf(In[nt][r] + bin + rr * (Hn[nt][r] + bhn));
      float hold = h[idx];
      float hnew = (1.0f - zz) * nn + zz * hold;
      h[idx] = hnew;
      h_half[idx] = (_Float16)hnew;
    }
  }
}

// ---------------------------------------------------------------------------
// Readout + correction: one wave per node.
// ---------------------------------------------------------------------------
__global__ void readout_kernel(const float* __restrict__ h,
                               const float* __restrict__ x,
                               const int* __restrict__ node_type,
                               const float* __restrict__ w1,
                               const float* __restrict__ b1,
                               const float* __restrict__ w2,
                               const float* __restrict__ b2,
                               float* __restrict__ out, int n) {
  int gid  = blockIdx.x * blockDim.x + threadIdx.x;
  int node = gid >> 5, lane = gid & 31;
  if (node >= n) return;
  int c0 = 2 * lane, c1 = c0 + 1;
  float t0 = b1[c0], t1 = b1[c1];
  const float* hrow = h + (size_t)node * H;
#pragma unroll 8
  for (int k = 0; k < H; ++k) {
    float hv = hrow[k];
    t0 += hv * w1[k * H + c0];
    t1 += hv * w1[k * H + c1];
  }
  t0 = fmaxf(t0, 0.0f);
  t1 = fmaxf(t1, 0.0f);
  float p = t0 * w2[c0] + t1 * w2[c1];
#pragma unroll
  for (int off = 16; off > 0; off >>= 1) p += __shfl_xor(p, off, 32);
  if (lane == 0) {
    float o = p + b2[0];
    out[node] = (node_type[node] == 0) ? (x[node * 4] + o) : 0.0f;
  }
}

// ---------------------------------------------------------------------------
extern "C" void kernel_launch(void* const* d_in, const int* in_sizes, int n_in,
                              void* d_out, int out_size, void* d_ws,
                              size_t ws_size, hipStream_t stream) {
  const float* x        = (const float*)d_in[0];
  const int*   node_ty  = (const int*)d_in[1];
  const int*   ei       = (const int*)d_in[2];
  const int*   etype    = (const int*)d_in[3];
  const float* in_w     = (const float*)d_in[4];
  const float* in_b     = (const float*)d_in[5];
  const float* ln_g     = (const float*)d_in[6];
  const float* ln_b     = (const float*)d_in[7];
  const float* mlp_w1   = (const float*)d_in[8];
  const float* mlp_b1   = (const float*)d_in[9];
  const float* mlp_w2   = (const float*)d_in[10];
  const float* mlp_b2   = (const float*)d_in[11];
  const float* gru_wi   = (const float*)d_in[12];
  const float* gru_wh   = (const float*)d_in[13];
  const float* gru_bi   = (const float*)d_in[14];
  const float* gru_bh   = (const float*)d_in[15];
  const float* ro_w1    = (const float*)d_in[16];
  const float* ro_b1    = (const float*)d_in[17];
  const float* ro_w2    = (const float*)d_in[18];
  const float* ro_b2    = (const float*)d_in[19];

  int nNodes = in_sizes[1];
  int nEdges = in_sizes[3];
  const int* esrc = ei;
  const int* edst = ei + nEdges;

  // workspace carve-up (all offsets 256B aligned for the fixed sizes)
  char* ws = (char*)d_ws;
  size_t off = 0;
  float* h = (float*)(ws + off);          off += (size_t)nNodes * H * sizeof(float);
  float* agg = (float*)(ws + off);        off += (size_t)nNodes * H * sizeof(float);
  _Float16* h_half = (_Float16*)(ws + off); off += (size_t)nNodes * H * sizeof(_Float16);
  _Float16* w1p = (_Float16*)(ws + off);  off += (size_t)NLAYERS * ETYPES * 16 * 512 * 2;
  _Float16* w2p = (_Float16*)(ws + off);  off += (size_t)NLAYERS * ETYPES * 8 * 512 * 2;
  _Float16* wip = (_Float16*)(ws + off);  off += (size_t)NLAYERS * 24 * 512 * 2;
  _Float16* whp = (_Float16*)(ws + off);  off += (size_t)NLAYERS * 24 * 512 * 2;
  (void)ws_size;

  // repack all weight matrices into f16 B-fragments
  {
    int tot = NLAYERS * ETYPES * 128 * 64;  // mlp_w1
    pack_b_kernel<<<(tot + 255) / 256, 256, 0, stream>>>(mlp_w1, w1p, 128, 64, tot);
    tot = NLAYERS * ETYPES * 64 * 64;       // mlp_w2
    pack_b_kernel<<<(tot + 255) / 256, 256, 0, stream>>>(mlp_w2, w2p, 64, 64, tot);
    tot = NLAYERS * 64 * 192;               // gru_wi
    pack_b_kernel<<<(tot + 255) / 256, 256, 0, stream>>>(gru_wi, wip, 64, 192, tot);
    pack_b_kernel<<<(tot + 255) / 256, 256, 0, stream>>>(gru_wh, whp, 64, 192, tot);
  }

  int nodeThreads = nNodes * 32;
  input_proj_kernel<<<(nodeThreads + 255) / 256, 256, 0, stream>>>(
      x, in_w, in_b, ln_g, ln_b, h, h_half, nNodes);

  int etiles = nEdges / 16;   // 2,000,000 / 16 = 125,000 exact
  int ntiles = nNodes / 16;   //   100,000 / 16 =   6,250 exact
  for (int l = 0; l < NLAYERS; ++l) {
    hipMemsetAsync(agg, 0, (size_t)nNodes * H * sizeof(float), stream);
    msg_kernel<<<(etiles + 7) / 8, 256, 0, stream>>>(
        h_half, esrc, edst, etype, w1p, w2p, mlp_b1, mlp_b2, agg, l, etiles);
    gru_kernel<<<(ntiles * 32 + 255) / 256, 256, 0, stream>>>(
        h, h_half, agg, wip, whp, gru_bi, gru_bh, l, ntiles);
  }

  readout_kernel<<<(nodeThreads + 255) / 256, 256, 0, stream>>>(
      h, x, node_ty, ro_w1, ro_b1, ro_w2, ro_b2, (float*)d_out, nNodes);
}
